// RegionProposalNetwork_86139864088942
// MI455X (gfx1250) — matrix operand order repaired and checked by reading, not compiled, not run on hardware
//
#include <hip/hip_runtime.h>

typedef __attribute__((ext_vector_type(8)))  float  v8f;
typedef __attribute__((ext_vector_type(16))) __bf16 v16bf;
typedef __attribute__((ext_vector_type(8)))  __bf16 v8bf;

#define TOT      159375
#define PRE_NMS  2000
#define POST_NMS 300
#define KTOT     4608   /* 512*9 */

static constexpr long OFF_LOCS   = 0;
static constexpr long OFF_SCORES = 2L * TOT * 4;                 // 1,275,000
static constexpr long OFF_ROIS   = OFF_SCORES + 2L * TOT * 2;    // 1,912,500
static constexpr long OFF_IDX    = OFF_ROIS + 2L * POST_NMS * 4; // 1,914,900
static constexpr long OFF_ANCH   = OFF_IDX + 2L * POST_NMS;      // 1,915,500

__device__ __forceinline__ __bf16 f2bf(float f) {
  unsigned u = __float_as_uint(f);
  unsigned r = u + 0x7FFFu + ((u >> 16) & 1u);   // round-to-nearest-even
  unsigned short h = (unsigned short)(r >> 16);
  __bf16 o; __builtin_memcpy(&o, &h, 2); return o;
}

__device__ __forceinline__ unsigned fkey(float f) {
  unsigned u = __float_as_uint(f);
  return (u & 0x80000000u) ? ~u : (u | 0x80000000u);
}

// ---------------- weight prep ----------------
__global__ void rpn_w1cast(const float* __restrict__ w, __bf16* __restrict__ o, int n) {
  int i = blockIdx.x * 256 + threadIdx.x;
  if (i < n) o[i] = f2bf(w[i]);
}

__global__ void rpn_headprep(const float* __restrict__ sw, const float* __restrict__ lw,
                             __bf16* __restrict__ o) {
  int i = blockIdx.x * 256 + threadIdx.x;
  if (i < 32 * 512) {
    int r = i >> 9, c = i & 511;
    float v = 0.f;
    if (r < 6)       v = sw[r * 512 + c];
    else if (r < 18) v = lw[(r - 6) * 512 + c];
    o[i] = f2bf(v);
  }
}

// ---- fused conv3x3 + ReLU + heads: implicit GEMM, 64px x 512ch tile, WMMA bf16 ----
// 8 waves: each wave covers ALL 64 pixels (4 A-frags) x a disjoint 64-channel slice
// (4 B-tiles) -> zero intra-workgroup B redundancy, 16 WMMA per 4 B loads.
__global__ __launch_bounds__(256)
void rpn_conv_head(const float* __restrict__ fm, const __bf16* __restrict__ W1,
                   const __bf16* __restrict__ Wh, const float* __restrict__ b1,
                   const float* __restrict__ sb, const float* __restrict__ lb,
                   float* __restrict__ out, int H, int W, int levelBase) {
  __shared__ __bf16 As[2][64][72];   // ping-pong: 64 pixels x 64 k (padded)
  __shared__ __bf16 Hs[64][520];     // hidden tile, 512 ch (padded)
  __shared__ int pixB[64], pixY[64], pixX[64];

  const int tid  = threadIdx.x;
  const int lane = tid & 31;
  const int wid  = tid >> 5;
  const int nbase = wid * 64;                 // this wave's channel slice
  const long tileBase = (long)blockIdx.x * 64;
  const int HW = H * W;
  const int totalPix = 2 * HW;

  // one-time per-tile pixel coordinates (no runtime divisions in hot loop)
  if (tid < 64) {
    long P = tileBase + tid;
    int bb = -1, y = 0, x = 0;
    if (P < totalPix) {
      bb = (int)(P / HW);
      int rem = (int)(P % HW);
      y = rem / W; x = rem % W;
    }
    pixB[tid] = bb; pixY[tid] = y; pixX[tid] = x;
  }
  __syncthreads();

  v8f vzero = {0.f, 0.f, 0.f, 0.f, 0.f, 0.f, 0.f, 0.f};
  v8f acc[4][4];                               // [m-tile][n-tile]
#pragma unroll
  for (int mi = 0; mi < 4; mi++)
#pragma unroll
    for (int t = 0; t < 4; t++) acc[mi][t] = vzero;

  const int kh   = (lane >> 4) * 16;    // A-layout half-select (K block of 16)
  const int kb16 = (lane >> 4) * 16;    // B-layout half-select (K block of 16)
  const int l15  = lane & 15;

  // this thread always stages pixel p = tid&63, k-slots kk = 4*i + (tid>>6)
  const int sp = tid & 63;
  const int skq = tid >> 6;                   // 0..3
  const int sB = pixB[sp], sY = pixY[sp], sX = pixX[sp];
  const long sbase = (sB >= 0) ? (long)sB * 512 * HW : 0;

  auto stage = [&](int buf, int kc) {
#pragma unroll
    for (int i = 0; i < 16; i++) {
      int kk = i * 4 + skq;                 // 0..63, each exactly once
      int k  = kc + kk;
      int c  = k / 9, t9 = k % 9;           // constant-divisor: cheap mulhi
      int y  = sY + t9 / 3 - 1;
      int x  = sX + t9 % 3 - 1;
      float v = 0.f;
      if (sB >= 0 && y >= 0 && y < H && x >= 0 && x < W)
        v = fm[sbase + (long)c * HW + (long)y * W + x];
      As[buf][sp][kk] = f2bf(v);
    }
  };

  stage(0, 0);
  __syncthreads();

  for (int kc = 0; kc < KTOT; kc += 64) {
    int buf = (kc >> 6) & 1;
    if (kc + 64 < KTOT) stage(buf ^ 1, kc + 64);   // overlap next-chunk staging

#pragma unroll
    for (int ks = 0; ks < 2; ks++) {
      int k0 = ks * 32;
      v16bf a[4];
#pragma unroll
      for (int mi = 0; mi < 4; mi++) {
        v8bf lo = *(const v8bf*)&As[buf][mi * 16 + l15][k0 + kh];
        v8bf hi = *(const v8bf*)&As[buf][mi * 16 + l15][k0 + kh + 8];
#pragma unroll
        for (int e = 0; e < 8; e++) { a[mi][e] = lo[e]; a[mi][e + 8] = hi[e]; }
      }
      int koff = kc + k0 + kb16;
#pragma unroll
      for (int t = 0; t < 4; t++) {
        int o = nbase + t * 16 + l15;
        v16bf bmat = *(const v16bf*)(W1 + (long)o * KTOT + koff);
#pragma unroll
        for (int mi = 0; mi < 4; mi++)
          acc[mi][t] = __builtin_amdgcn_wmma_f32_16x16x32_bf16(
              false, a[mi], false, bmat, (short)0, acc[mi][t], false, false);
      }
    }
    __syncthreads();   // readers of As[buf] done; writers of As[buf^1] done
  }

  // ---- bias + ReLU, stash hidden tile (bf16) in LDS ----
  const int mofs = (lane >> 4) * 8;
#pragma unroll
  for (int t = 0; t < 4; t++) {
    int ch = nbase + t * 16 + l15;
    float bias = b1[ch];
#pragma unroll
    for (int mi = 0; mi < 4; mi++) {
#pragma unroll
      for (int r = 0; r < 8; r++) {
        int p = mi * 16 + r + mofs;
        Hs[p][ch] = f2bf(fmaxf(acc[mi][t][r] + bias, 0.f));
      }
    }
  }
  __syncthreads();

  // ---- head GEMM: 64 pixels x 32 out (18 used) x K=512, waves 0..3 ----
  if (wid < 4) {
    int pb = wid * 16;
    int mr2 = pb + l15;
    v8f h0 = vzero, h1 = vzero;
    for (int kk = 0; kk < 512; kk += 32) {
      v8bf lo = *(const v8bf*)&Hs[mr2][kk + kh];
      v8bf hi = *(const v8bf*)&Hs[mr2][kk + kh + 8];
      v16bf a;
#pragma unroll
      for (int e = 0; e < 8; e++) { a[e] = lo[e]; a[e + 8] = hi[e]; }
      int koff = kk + kb16;
      v16bf bm0 = *(const v16bf*)(Wh + (long)l15 * 512 + koff);
      v16bf bm1 = *(const v16bf*)(Wh + (long)(16 + l15) * 512 + koff);
      h0 = __builtin_amdgcn_wmma_f32_16x16x32_bf16(false, a, false, bm0, (short)0, h0, false, false);
      h1 = __builtin_amdgcn_wmma_f32_16x16x32_bf16(false, a, false, bm1, (short)0, h1, false, false);
    }
#pragma unroll
    for (int half = 0; half < 2; half++) {
      v8f hv = half ? h1 : h0;
      int ch = half * 16 + l15;
#pragma unroll
      for (int r = 0; r < 8; r++) {
        int lp = pb + r + mofs;                 // local pixel 0..63
        int bb = pixB[lp];
        if (ch < 18 && bb >= 0) {
          float v = hv[r] + (ch < 6 ? sb[ch] : lb[ch - 6]);
          int pos = pixY[lp] * W + pixX[lp];
          if (ch < 6) {
            int aR = ch >> 1, s = ch & 1;
            long ai = (long)levelBase + (long)pos * 3 + aR;
            out[OFF_SCORES + ((long)bb * TOT + ai) * 2 + s] = v;
          } else {
            int j = ch - 6, aR = j >> 2, jj = j & 3;
            long ai = (long)levelBase + (long)pos * 3 + aR;
            out[OFF_LOCS + ((long)bb * TOT + ai) * 4 + jj] = v;
          }
        }
      }
    }
  }
}

// ---------------- anchors + softmax-fg + box decode/clip/mask ----------------
__global__ __launch_bounds__(256)
void rpn_decode(float* __restrict__ out, float* __restrict__ boxes,
                float* __restrict__ scm, const int* __restrict__ pH,
                const int* __restrict__ pW) {
  long i = (long)blockIdx.x * 256 + threadIdx.x;
  if (i >= 2L * TOT) return;
  int b = (int)(i / TOT), aidx = (int)(i % TOT);
  int base, Hh, stride; float scale;
  if (aidx < 120000)      { base = 0;      Hh = 200; stride = 4;  scale = 4.f;  }
  else if (aidx < 150000) { base = 120000; Hh = 100; stride = 8;  scale = 8.f;  }
  else if (aidx < 157500) { base = 150000; Hh = 50;  stride = 16; scale = 16.f; }
  else                    { base = 157500; Hh = 25;  stride = 32; scale = 32.f; }
  int rem = aidx - base;
  int pos = rem / 3, rr = rem % 3;
  int y = pos / Hh, x = pos % Hh;
  float ratio = (rr == 0) ? 0.5f : ((rr == 1) ? 1.f : 2.f);
  float hh = 16.f * scale * sqrtf(ratio);
  float ww = 16.f * scale * sqrtf(1.f / ratio);
  float ay1 = (float)(y * stride) + 8.f - 0.5f * hh;
  float ax1 = (float)(x * stride) + 8.f - 0.5f * ww;
  float ay2 = ay1 + hh, ax2 = ax1 + ww;
  if (b == 0) {
    out[OFF_ANCH + (long)aidx * 4 + 0] = ay1;
    out[OFF_ANCH + (long)aidx * 4 + 1] = ax1;
    out[OFF_ANCH + (long)aidx * 4 + 2] = ay2;
    out[OFF_ANCH + (long)aidx * 4 + 3] = ax2;
  }
  float acy = ay1 + 0.5f * hh, acx = ax1 + 0.5f * ww;
  const float* lp = out + OFF_LOCS + i * 4;
  float cy = lp[0] * hh + acy, cx = lp[1] * ww + acx;
  float dh = expf(lp[2]) * hh, dw = expf(lp[3]) * ww;
  float imH = (float)pH[0], imW = (float)pW[0];
  float y1 = fminf(fmaxf(cy - 0.5f * dh, 0.f), imH);
  float x1 = fminf(fmaxf(cx - 0.5f * dw, 0.f), imW);
  float y2 = fminf(fmaxf(cy + 0.5f * dh, 0.f), imH);
  float x2 = fminf(fmaxf(cx + 0.5f * dw, 0.f), imW);
  bool valid = ((y2 - y1) >= 16.f) && ((x2 - x1) >= 16.f);
  float s0 = out[OFF_SCORES + i * 2], s1 = out[OFF_SCORES + i * 2 + 1];
  float fg = 1.f / (1.f + expf(s0 - s1));
  float* bp = boxes + i * 4;
  bp[0] = y1; bp[1] = x1; bp[2] = y2; bp[3] = x2;
  scm[i] = valid ? fg : -__builtin_inff();
}

// ---------------- top-k 2000 per batch: radix select + bitonic sort --------
__global__ __launch_bounds__(1024)
void rpn_topk(const float* __restrict__ scm, int* __restrict__ tIdx,
              unsigned* __restrict__ tKey) {
  int b = blockIdx.x, tid = threadIdx.x;
  const float* sc = scm + (long)b * TOT;
  __shared__ unsigned hist[256];
  __shared__ int sByte, sAbove, cntG, cntE;
  __shared__ unsigned long long comp[2048];

  unsigned prefix = 0, pmask = 0; int remK = PRE_NMS;
  for (int shift = 24; shift >= 0; shift -= 8) {
    for (int i = tid; i < 256; i += 1024) hist[i] = 0;
    __syncthreads();
    for (int i = tid; i < TOT; i += 1024) {
      unsigned k = fkey(sc[i]);
      if ((k & pmask) == prefix) atomicAdd(&hist[(k >> shift) & 255], 1u);
    }
    __syncthreads();
    if (tid == 0) {
      int c = 0, chosen = 0;
      for (int v = 255; v >= 0; --v) {
        int h = (int)hist[v];
        if (c + h >= remK) { chosen = v; break; }
        c += h;
      }
      sByte = chosen; sAbove = c;
    }
    __syncthreads();
    prefix |= ((unsigned)sByte) << shift;
    pmask  |= (0xFFu << shift);
    remK   -= sAbove;
    __syncthreads();
  }
  unsigned T = prefix;
  if (tid == 0) { cntG = 0; cntE = 0; }
  __syncthreads();
  for (int i = tid; i < TOT; i += 1024) {
    unsigned k = fkey(sc[i]);
    if (k > T) {
      int p = atomicAdd(&cntG, 1);
      if (p < PRE_NMS) comp[p] = ((unsigned long long)k << 32) | (unsigned)(~i);
    }
  }
  __syncthreads();
  int g = cntG;
  for (int i = tid; i < TOT; i += 1024) {
    unsigned k = fkey(sc[i]);
    if (k == T) {
      int p = atomicAdd(&cntE, 1);
      if (g + p < PRE_NMS) comp[g + p] = ((unsigned long long)k << 32) | (unsigned)(~i);
    }
  }
  __syncthreads();
  for (int i = tid; i < 2048; i += 1024) if (i >= PRE_NMS) comp[i] = 0ull;
  __syncthreads();
  // bitonic sort, descending (key desc, then index asc via ~idx)
  for (int kk = 2; kk <= 2048; kk <<= 1) {
    for (int j = kk >> 1; j > 0; j >>= 1) {
      for (int t = tid; t < 2048; t += 1024) {
        int ixj = t ^ j;
        if (ixj > t) {
          bool descBlock = ((t & kk) == 0);
          unsigned long long a = comp[t], bb = comp[ixj];
          bool doSwap = descBlock ? (a < bb) : (a > bb);
          if (doSwap) { comp[t] = bb; comp[ixj] = a; }
        }
      }
      __syncthreads();
    }
  }
  for (int i = tid; i < PRE_NMS; i += 1024) {
    unsigned long long c = comp[i];
    tKey[(long)b * PRE_NMS + i] = (unsigned)(c >> 32);
    tIdx[(long)b * PRE_NMS + i] = (int)(~(unsigned)(c & 0xFFFFFFFFu));
  }
}

// ---------------- greedy NMS + ROI compaction ----------------
__global__ __launch_bounds__(1024)
void rpn_nms(const float* __restrict__ boxes, const int* __restrict__ tIdx,
             const unsigned* __restrict__ tKey, float* __restrict__ out) {
  int b = blockIdx.x, tid = threadIdx.x;
  __shared__ float bx[PRE_NMS][4];
  __shared__ int keep[PRE_NMS];
  __shared__ int order[POST_NMS];
  __shared__ int sCnt;
  for (int i = tid; i < PRE_NMS; i += 1024) {
    int gi = tIdx[(long)b * PRE_NMS + i];
    const float* p = boxes + ((long)b * TOT + gi) * 4;
    bx[i][0] = p[0]; bx[i][1] = p[1]; bx[i][2] = p[2]; bx[i][3] = p[3];
    keep[i] = tKey[(long)b * PRE_NMS + i] > 0x007FFFFFu; // score > -inf
  }
  __syncthreads();
  for (int i = 0; i < PRE_NMS; i++) {
    if (keep[i]) {
      float y1 = bx[i][0], x1 = bx[i][1], y2 = bx[i][2], x2 = bx[i][3];
      float ai = (y2 - y1) * (x2 - x1);
      for (int j = i + 1 + tid; j < PRE_NMS; j += 1024) {
        if (keep[j]) {
          float iy = fmaxf(0.f, fminf(y2, bx[j][2]) - fmaxf(y1, bx[j][0]));
          float ix = fmaxf(0.f, fminf(x2, bx[j][3]) - fmaxf(x1, bx[j][1]));
          float inter = iy * ix;
          float aj = (bx[j][2] - bx[j][0]) * (bx[j][3] - bx[j][1]);
          float iou = inter / (ai + aj - inter + 1e-10f);
          if (iou > 0.7f) keep[j] = 0;
        }
      }
    }
    __syncthreads();
  }
  if (tid == 0) {
    int c = 0;
    for (int i = 0; i < PRE_NMS && c < POST_NMS; i++)
      if (keep[i]) order[c++] = i;
    sCnt = c;
  }
  __syncthreads();
  for (int r = tid; r < POST_NMS; r += 1024) {
    float v0 = 0.f, v1 = 0.f, v2 = 0.f, v3 = 0.f;
    if (r < sCnt) { int i = order[r]; v0 = bx[i][0]; v1 = bx[i][1]; v2 = bx[i][2]; v3 = bx[i][3]; }
    long o = OFF_ROIS + ((long)b * POST_NMS + r) * 4;
    out[o + 0] = v0; out[o + 1] = v1; out[o + 2] = v2; out[o + 3] = v3;
    out[OFF_IDX + (long)b * POST_NMS + r] = (float)b;
  }
}

// ---------------- launcher ----------------
extern "C" void kernel_launch(void* const* d_in, const int* in_sizes, int n_in,
                              void* d_out, int out_size, void* d_ws, size_t ws_size,
                              hipStream_t stream) {
  const float* fm[4] = {(const float*)d_in[0], (const float*)d_in[1],
                        (const float*)d_in[2], (const float*)d_in[3]};
  const float* conv1_w = (const float*)d_in[4];
  const float* conv1_b = (const float*)d_in[5];
  const float* score_w = (const float*)d_in[6];
  const float* score_b = (const float*)d_in[7];
  const float* loc_w   = (const float*)d_in[8];
  const float* loc_b   = (const float*)d_in[9];
  const int* imgH = (const int*)d_in[10];
  const int* imgW = (const int*)d_in[11];
  float* out = (float*)d_out;

  char* ws = (char*)d_ws;
  auto al = [](size_t x) { return (x + 255) & ~(size_t)255; };
  size_t off = 0;
  __bf16* W1bf = (__bf16*)(ws + off);   off += al((size_t)512 * KTOT * 2);
  __bf16* Whbf = (__bf16*)(ws + off);   off += al((size_t)32 * 512 * 2);
  float* boxes = (float*)(ws + off);    off += al((size_t)2 * TOT * 4 * 4);
  float* scm   = (float*)(ws + off);    off += al((size_t)2 * TOT * 4);
  int* tIdx    = (int*)(ws + off);      off += al((size_t)2 * PRE_NMS * 4);
  unsigned* tKey = (unsigned*)(ws + off);

  // 1) weight prep
  rpn_w1cast<<<(512 * KTOT + 255) / 256, 256, 0, stream>>>(conv1_w, W1bf, 512 * KTOT);
  rpn_headprep<<<(32 * 512 + 255) / 256, 256, 0, stream>>>(score_w, loc_w, Whbf);

  // 2) fused conv + heads per pyramid level (64-pixel tiles)
  const int Hs[4]   = {200, 100, 50, 25};
  const int base[4] = {0, 120000, 150000, 157500};
  for (int l = 0; l < 4; l++) {
    int H = Hs[l];
    int tiles = (2 * H * H + 63) / 64;
    rpn_conv_head<<<tiles, 256, 0, stream>>>(fm[l], W1bf, Whbf, conv1_b, score_b,
                                             loc_b, out, H, H, base[l]);
  }

  // 3) anchors + fg + decode/clip/mask
  rpn_decode<<<(int)((2L * TOT + 255) / 256), 256, 0, stream>>>(out, boxes, scm, imgH, imgW);

  // 4) top-k 2000 per batch
  rpn_topk<<<2, 1024, 0, stream>>>(scm, tIdx, tKey);

  // 5) greedy NMS + ROI output
  rpn_nms<<<2, 1024, 0, stream>>>(boxes, tIdx, tKey, out);
}
